// SeedGNN_32847909880048
// MI455X (gfx1250) — matrix operand order repaired and checked by reading, not compiled, not run on hardware
//
#include <hip/hip_runtime.h>

#define NN    768
#define NPIX  (768 * 768)
#define NSEEDS 50
#define HID   32
#define NLAYERS 6

typedef __bf16 bf16_t;
typedef __attribute__((ext_vector_type(16))) __bf16 v16bf;
typedef __attribute__((ext_vector_type(8)))  __bf16 v8bf;
typedef __attribute__((ext_vector_type(8)))  float  v8f;

__device__ __forceinline__ unsigned short f2bf(float f) {
    unsigned u = __float_as_uint(f);
    unsigned r = 0x7FFFu + ((u >> 16) & 1u);   // round-to-nearest-even
    return (unsigned short)((u + r) >> 16);
}

// ---------------------------------------------------------------------------
// Batched GEMM: C[z] = A[z] * B[z], A row-major [M,K] bf16, Bt = B^T row-major
// [N,K] bf16 (i.e. B column-major), C row-major [M,N] (bf16 or f32).
// 8 waves / block; wave tile 32x32 (2x2 WMMA 16x16x32 bf16); block tile 64x128.
// ---------------------------------------------------------------------------
template<bool OUT_BF16>
__global__ __launch_bounds__(256)
void k_gemm(const unsigned short* __restrict__ A,
            const unsigned short* __restrict__ Bt,
            void* __restrict__ Cv,
            long strideA, long strideB, long strideC)
{
    const int  N    = NN;
    const long z    = blockIdx.z;
    const bf16_t* Ab  = (const bf16_t*)(A  + z * strideA);
    const bf16_t* Btb = (const bf16_t*)(Bt + z * strideB);

    const int tid  = threadIdx.x;
    const int wave = tid >> 5;
    const int lane = tid & 31;
    const int half = lane >> 4;     // which 16-lane half of the wave
    const int r    = lane & 15;     // row (A) / column (B) within tile

    const int m0 = blockIdx.y * 64  + (wave & 1) * 32;
    const int n0 = blockIdx.x * 128 + (wave >> 1) * 32;

    v8f acc[2][2] = {};

    for (int k0 = 0; k0 < N; k0 += 32) {
        v16bf afrag[2], bfrag[2];
        // A fragment: lane(half,r) holds row m0+r, K = {half*8..half*8+7} u {16+half*8..}
#pragma unroll
        for (int i = 0; i < 2; ++i) {
            const bf16_t* p = Ab + (long)(m0 + i * 16 + r) * N + k0 + half * 8;
            __builtin_prefetch(p + 32, 0, 0);
            v16bf f;
            ((v8bf*)&f)[0] = *(const v8bf*)p;
            ((v8bf*)&f)[1] = *(const v8bf*)(p + 16);
            afrag[i] = f;
        }
        // B fragment: lane(half,r) holds column n0+r, K = {half*8..half*8+7} u {16+half*8..}
#pragma unroll
        for (int j = 0; j < 2; ++j) {
            const bf16_t* p = Btb + (long)(n0 + j * 16 + r) * N + k0 + half * 8;
            __builtin_prefetch(p + 32, 0, 0);
            v16bf f;
            ((v8bf*)&f)[0] = *(const v8bf*)p;
            ((v8bf*)&f)[1] = *(const v8bf*)(p + 16);
            bfrag[j] = f;
        }
#pragma unroll
        for (int i = 0; i < 2; ++i)
#pragma unroll
            for (int j = 0; j < 2; ++j)
                acc[i][j] = __builtin_amdgcn_wmma_f32_16x16x32_bf16(
                    false, afrag[i], false, bfrag[j],
                    (short)0, acc[i][j], false, false);
    }

    // C/D layout: VGPR d, lane(half,n): M = half*8 + d, N = n
    if (OUT_BF16) {
        unsigned short* C = (unsigned short*)Cv + z * strideC;
#pragma unroll
        for (int i = 0; i < 2; ++i)
#pragma unroll
            for (int j = 0; j < 2; ++j) {
                const int mb = m0 + i * 16 + half * 8;
                const int nb = n0 + j * 16 + r;
#pragma unroll
                for (int d = 0; d < 8; ++d)
                    C[(long)(mb + d) * N + nb] = f2bf(acc[i][j][d]);
            }
    } else {
        float* C = (float*)Cv + z * strideC;
#pragma unroll
        for (int i = 0; i < 2; ++i)
#pragma unroll
            for (int j = 0; j < 2; ++j) {
                const int mb = m0 + i * 16 + half * 8;
                const int nb = n0 + j * 16 + r;
#pragma unroll
                for (int d = 0; d < 8; ++d)
                    C[(long)(mb + d) * N + nb] = acc[i][j][d];
            }
    }
}

// ---------------------------------------------------------------------------
// Small helper kernels
// ---------------------------------------------------------------------------
__global__ void k_f32_to_bf16(const float* __restrict__ in,
                              unsigned short* __restrict__ out, long n) {
    long i = blockIdx.x * (long)blockDim.x + threadIdx.x;
    if (i < n) out[i] = f2bf(in[i]);
}

__global__ void k_zero32(unsigned* __restrict__ p, long n) {
    long i = blockIdx.x * (long)blockDim.x + threadIdx.x;
    if (i < n) p[i] = 0u;
}

// S^T channel 0: S[j,b] one-hot at (s0[k], s1[k])  ->  S_T[b*NN + j] = 1.0bf
__global__ void k_scatter_S0(const int* __restrict__ seeds,
                             unsigned short* __restrict__ S_T) {
    int k = threadIdx.x;
    if (k < NSEEDS) {
        int j = seeds[k];
        int b = seeds[NSEEDS + k];
        S_T[(long)b * NN + j] = 0x3F80;    // bf16(1.0)
    }
}

__global__ void k_seed_flags(const int* __restrict__ seeds,
                             int* __restrict__ rowf, int* __restrict__ colf) {
    int t = threadIdx.x;          // blockDim == NN
    rowf[t] = 0; colf[t] = 0;
    __syncthreads();
    if (t < NSEEDS) {
        rowf[seeds[t]] = 1;
        colf[seeds[NSEEDS + t]] = 1;
    }
}

// Match[i,c] = sum_h H[h][i,c] * roW[h] + rob
__global__ void k_match(const float* __restrict__ H, int HC,
                        const float* __restrict__ roW, const float* __restrict__ rob,
                        float* __restrict__ Match) {
    long i = blockIdx.x * (long)blockDim.x + threadIdx.x;
    if (i >= NPIX) return;
    float s = rob[0];
    for (int h = 0; h < HC; ++h) s += H[(long)h * NPIX + i] * roW[h];
    Match[i] = s;
}

__global__ __launch_bounds__(256)
void k_row_stats(const float* __restrict__ Match, float* __restrict__ rmax,
                 float* __restrict__ rsum) {
    __shared__ float sm[256];
    const int row = blockIdx.x, t = threadIdx.x;
    const float* p = Match + (long)row * NN;
    float m = -3.4e38f;
    for (int c = t; c < NN; c += 256) m = fmaxf(m, p[c]);
    sm[t] = m; __syncthreads();
    for (int s = 128; s; s >>= 1) { if (t < s) sm[t] = fmaxf(sm[t], sm[t + s]); __syncthreads(); }
    const float rm = sm[0]; __syncthreads();
    float su = 0.f;
    for (int c = t; c < NN; c += 256) su += __expf(p[c] - rm);
    sm[t] = su; __syncthreads();
    for (int s = 128; s; s >>= 1) { if (t < s) sm[t] += sm[t + s]; __syncthreads(); }
    if (t == 0) { rmax[row] = rm; rsum[row] = sm[0]; }
}

__global__ __launch_bounds__(256)
void k_col_stats(const float* __restrict__ Match, float* __restrict__ cmax,
                 float* __restrict__ csum) {
    __shared__ float sm[256];
    const int col = blockIdx.x, t = threadIdx.x;
    float m = -3.4e38f;
    for (int r = t; r < NN; r += 256) m = fmaxf(m, Match[(long)r * NN + col]);
    sm[t] = m; __syncthreads();
    for (int s = 128; s; s >>= 1) { if (t < s) sm[t] = fmaxf(sm[t], sm[t + s]); __syncthreads(); }
    const float cm = sm[0]; __syncthreads();
    float su = 0.f;
    for (int r = t; r < NN; r += 256) su += __expf(Match[(long)r * NN + col] - cm);
    sm[t] = su; __syncthreads();
    for (int s = 128; s; s >>= 1) { if (t < s) sm[t] += sm[t + s]; __syncthreads(); }
    if (t == 0) { cmax[col] = cm; csum[col] = sm[0]; }
}

// M = 0.5*(softmax_row + softmax_col), zero seed rows/cols
__global__ void k_combine(const float* __restrict__ Match,
                          const float* __restrict__ rmax, const float* __restrict__ rsum,
                          const float* __restrict__ cmax, const float* __restrict__ csum,
                          const int* __restrict__ rowf, const int* __restrict__ colf,
                          float* __restrict__ M) {
    long i = blockIdx.x * (long)blockDim.x + threadIdx.x;
    if (i >= NPIX) return;
    const int a = (int)(i / NN), b = (int)(i % NN);
    const float s = Match[i];
    float v = 0.5f * (__expf(s - rmax[a]) / rsum[a] + __expf(s - cmax[b]) / csum[b]);
    if (rowf[a] | colf[b]) v = 0.f;
    M[i] = v;
}

__global__ void k_set_ones(const int* __restrict__ seeds, float* __restrict__ M) {
    int k = threadIdx.x;
    if (k < NSEEDS) M[(long)seeds[k] * NN + seeds[NSEEDS + k]] = 1.0f;
}

// ---------------------------------------------------------------------------
// Auction assignment (replaces CPU Hungarian): maximize sum_i M[i, col[i]].
// Single workgroup of NN threads; bids resolved with 64-bit atomicMax
// (positive-float bits in high word, bidder row in low word).
// ---------------------------------------------------------------------------
__global__ __launch_bounds__(NN)
void k_auction(const float* __restrict__ M, float* __restrict__ price,
               int* __restrict__ colOf, int* __restrict__ owner,
               unsigned long long* __restrict__ bids) {
    const int t = threadIdx.x;
    price[t] = 0.f; colOf[t] = -1; owner[t] = -1;
    __shared__ int s_unassigned;
    __syncthreads();
    const float eps = 1e-4f;
    for (int iter = 0; iter < 6000; ++iter) {
        bids[t] = 0ULL;
        if (t == 0) s_unassigned = 0;
        __syncthreads();
        if (colOf[t] < 0) {                    // row t bids
            atomicAdd(&s_unassigned, 1);
            const float* row = M + (long)t * NN;
            float v1 = -3.4e38f, v2 = -3.4e38f; int c1 = 0;
            for (int c = 0; c < NN; ++c) {
                const float v = row[c] - price[c];
                if (v > v1) { v2 = v1; v1 = v; c1 = c; }
                else if (v > v2) v2 = v;
            }
            const float bid = price[c1] + (v1 - v2) + eps;   // > 0
            const unsigned long long enc =
                ((unsigned long long)__float_as_uint(bid) << 32) | (unsigned)t;
            atomicMax(&bids[c1], enc);
        }
        __syncthreads();
        if (s_unassigned == 0) break;
        const unsigned long long e = bids[t];  // column t resolves
        if (e) {
            const int rwin = (int)(e & 0xFFFFFFFFULL);
            const int old  = owner[t];
            if (old >= 0) colOf[old] = -1;
            owner[t] = rwin; colOf[rwin] = t;
            price[t] = __uint_as_float((unsigned)(e >> 32));
        }
        __syncthreads();
    }
}

// ---------------------------------------------------------------------------
// Next-layer S (stored transposed per channel): ch 0..30 = relu(H*W+b)/1000,
// ch 31 = M * (10 at (j, colOf[j])).
// ---------------------------------------------------------------------------
template<int HC>
__global__ __launch_bounds__(256)
void k_build_S(const float* __restrict__ H, const float* __restrict__ W,
               const float* __restrict__ bW, const float* __restrict__ M,
               const int* __restrict__ colOf, unsigned short* __restrict__ S_T) {
    long i = blockIdx.x * (long)blockDim.x + threadIdx.x;
    if (i >= NPIX) return;
    const int j = (int)(i / NN), b = (int)(i % NN);
    const long tpos = (long)b * NN + j;
    float h[HC];
#pragma unroll
    for (int t = 0; t < HC; ++t) h[t] = H[(long)t * NPIX + i];
#pragma unroll
    for (int k = 0; k < HID - 1; ++k) {
        float s = bW[k];
#pragma unroll
        for (int t = 0; t < HC; ++t) s += h[t] * W[t * (HID - 1) + k];
        s = fmaxf(s, 0.f) * 1e-3f;
        S_T[(long)k * NPIX + tpos] = f2bf(s);
    }
    const float z = (colOf[j] == b) ? (M[i] * 10.f) : 0.f;
    S_T[(long)(HID - 1) * NPIX + tpos] = f2bf(z);
}

__global__ void k_copy(const float* __restrict__ src, float* __restrict__ dst) {
    long i = blockIdx.x * (long)blockDim.x + threadIdx.x;
    if (i < NPIX) dst[i] = src[i];
}

// ---------------------------------------------------------------------------
extern "C" void kernel_launch(void* const* d_in, const int* in_sizes, int n_in,
                              void* d_out, int out_size, void* d_ws, size_t ws_size,
                              hipStream_t stream)
{
    (void)in_sizes; (void)n_in; (void)out_size; (void)ws_size;
    const float* G1    = (const float*)d_in[0];
    const float* G2    = (const float*)d_in[1];
    const int*   seeds = (const int*)d_in[2];
    const float *mlp_W[NLAYERS], *mlp_b[NLAYERS], *ro_W[NLAYERS], *ro_b[NLAYERS];
    for (int l = 0; l < NLAYERS; ++l) {
        mlp_W[l] = (const float*)d_in[3 + l];
        mlp_b[l] = (const float*)d_in[3 + NLAYERS + l];
        ro_W[l]  = (const float*)d_in[3 + 2 * NLAYERS + l];
        ro_b[l]  = (const float*)d_in[3 + 3 * NLAYERS + l];
    }

    char* wp = (char*)d_ws;
    auto alloc = [&](size_t bytes) -> char* {
        char* p = wp; wp += (bytes + 255) & ~(size_t)255; return p;
    };
    unsigned short* G1b = (unsigned short*)alloc((size_t)NPIX * 2);
    unsigned short* G2b = (unsigned short*)alloc((size_t)NPIX * 2);
    unsigned short* S   = (unsigned short*)alloc((size_t)HID * NPIX * 2); // S^T per channel
    unsigned short* T   = (unsigned short*)alloc((size_t)HID * NPIX * 2); // row-major per channel
    float* H     = (float*)alloc((size_t)HID * NPIX * 4);
    float* Match = (float*)alloc((size_t)NPIX * 4);
    float* rmax  = (float*)alloc(NN * 4);
    float* rsum  = (float*)alloc(NN * 4);
    float* cmax  = (float*)alloc(NN * 4);
    float* csum  = (float*)alloc(NN * 4);
    int*   rowf  = (int*)alloc(NN * 4);
    int*   colf  = (int*)alloc(NN * 4);
    float* price = (float*)alloc(NN * 4);
    int*   colOf = (int*)alloc(NN * 4);
    int*   owner = (int*)alloc(NN * 4);
    unsigned long long* bids = (unsigned long long*)alloc(NN * 8);

    float* out = (float*)d_out;
    const int PB = (NPIX + 255) / 256;

    // Prep: bf16 copies of G1/G2; S^T layer-0 channel = seed one-hot; seed flags.
    k_f32_to_bf16<<<PB, 256, 0, stream>>>(G1, G1b, NPIX);
    k_f32_to_bf16<<<PB, 256, 0, stream>>>(G2, G2b, NPIX);
    k_zero32<<<(NPIX / 2 + 255) / 256, 256, 0, stream>>>((unsigned*)S, NPIX / 2);
    k_scatter_S0<<<1, 64, 0, stream>>>(seeds, S);
    k_seed_flags<<<1, NN, 0, stream>>>(seeds, rowf, colf);

    for (int l = 0; l < NLAYERS; ++l) {
        const int HC = (l == 0) ? 1 : HID;
        dim3 g(NN / 128, NN / 64, HC);
        // T_h = G1 @ S_h        (A = G1 row-major; Bt = S^T[h]; out bf16)
        k_gemm<true ><<<g, 256, 0, stream>>>(G1b, S, (void*)T, 0, NPIX, NPIX);
        // H_h = T_h @ G2        (A = T[h]; Bt = G2 (symmetric); out f32)
        k_gemm<false><<<g, 256, 0, stream>>>(T, G2b, (void*)H, NPIX, 0, NPIX);

        k_match<<<PB, 256, 0, stream>>>(H, HC, ro_W[l], ro_b[l], Match);
        k_row_stats<<<NN, 256, 0, stream>>>(Match, rmax, rsum);
        k_col_stats<<<NN, 256, 0, stream>>>(Match, cmax, csum);

        float* Ml = out + (size_t)(1 + l) * NPIX;          // Ys[l]
        k_combine<<<PB, 256, 0, stream>>>(Match, rmax, rsum, cmax, csum, rowf, colf, Ml);
        k_set_ones<<<1, 64, 0, stream>>>(seeds, Ml);

        if (l < NLAYERS - 1) {
            k_auction<<<1, NN, 0, stream>>>(Ml, price, colOf, owner, bids);
            if (l == 0)
                k_build_S<1  ><<<PB, 256, 0, stream>>>(H, mlp_W[l], mlp_b[l], Ml, colOf, S);
            else
                k_build_S<HID><<<PB, 256, 0, stream>>>(H, mlp_W[l], mlp_b[l], Ml, colOf, S);
        }
    }
    // Output 0: Ys[-1]
    k_copy<<<PB, 256, 0, stream>>>(out + (size_t)NLAYERS * NPIX, out);
}
